// DiffGraphTransformerSBM_17703855194483
// MI455X (gfx1250) — matrix operand order repaired and checked by reading, not compiled, not run on hardware
//
#include <hip/hip_runtime.h>

// ---------------------------------------------------------------------------
// Problem constants (from reference)
// ---------------------------------------------------------------------------
#define BATCH 32
#define NN    512          // nodes
#define INF_  64           // input feature dim
#define DMODEL 512
#define NHEAD 8
#define DH    64           // DMODEL / NHEAD
#define DFF   2048
#define NLAYER 4
#define NCLS  6
#define MROWS (BATCH * NN) // 16384
#define TD    (3 * DMODEL) // 1536

typedef __attribute__((ext_vector_type(16))) __bf16 v16bf;
typedef __attribute__((ext_vector_type(8)))  __bf16 v8bf;
typedef __attribute__((ext_vector_type(8)))  float  v8f;

// ---------------------------------------------------------------------------
// bf16 fragment loads (operands pre-converted in memory)
//
// A-matrix 16x32 (MxK) bf16 order used by this code (contiguous k, not the ISA table order):
//   lane 0-15 : M = lane,    K = {0..15}
//   lane 16-31: M = lane-16, K = {16..31}
// B-matrix 32x16 (KxN) bf16 layout:
//   lane holds column N = lane&15, 16 consecutive K starting at (lane>>4)*16
// C/D 16x16 f32: VGPR r -> row = r + (lane>>4)*8, col = lane&15
// ---------------------------------------------------------------------------

__device__ __forceinline__ v16bf frag_a_bf16(const __bf16* A, int lda,
                                             int row0, int k0, int lane) {
    const int m  = row0 + (lane & 15);
    const int kb = k0 + ((lane >> 4) << 4);    // +0 or +16
    const __bf16* p = A + (size_t)m * lda + kb;
    v8bf lo = *reinterpret_cast<const v8bf*>(p);        // K kb..kb+7
    v8bf hi = *reinterpret_cast<const v8bf*>(p + 8);    // K kb+8..kb+15
    return __builtin_shufflevector(lo, hi, 0,1,2,3,4,5,6,7,8,9,10,11,12,13,14,15);
}

// W is (N,K) row-major bf16; row n of W == column n of the KxN B operand.
__device__ __forceinline__ v16bf frag_b_bf16(const __bf16* W, int ldk,
                                             int n0, int k0, int lane) {
    const int n  = n0 + (lane & 15);
    const int kb = k0 + ((lane >> 4) << 4);    // +0 or +16
    return *reinterpret_cast<const v16bf*>(W + (size_t)n * ldk + kb);
}

__device__ __forceinline__ v16bf frag_b_bf16_guard(const __bf16* W, int ldk,
                                                   int n0, int k0, int nLim, int lane) {
    const int n = n0 + (lane & 15);
    if (n < nLim) return frag_b_bf16(W, ldk, n0, k0, lane);
    v16bf f;
    #pragma unroll
    for (int i = 0; i < 16; ++i) f[i] = (__bf16)0.0f;
    return f;
}

__device__ __forceinline__ v8f wmma_bf16(v16bf a, v16bf b, v8f c) {
    return __builtin_amdgcn_wmma_f32_16x16x32_bf16(false, a, false, b,
                                                   (short)0, c, false, false);
}

// ---------------------------------------------------------------------------
// f32 -> bf16 pack (8 elements / thread); count must be a multiple of 8.
// ---------------------------------------------------------------------------
__global__ __launch_bounds__(256)
void cvt_bf16_kernel(const float* __restrict__ src, __bf16* __restrict__ dst, int n8) {
    const int i = blockIdx.x * 256 + threadIdx.x;
    if (i >= n8) return;
    const float4* s = reinterpret_cast<const float4*>(src) + (size_t)i * 2;
    float4 u = s[0], v = s[1];
    v8bf o;
    o[0] = (__bf16)u.x; o[1] = (__bf16)u.y; o[2] = (__bf16)u.z; o[3] = (__bf16)u.w;
    o[4] = (__bf16)v.x; o[5] = (__bf16)v.y; o[6] = (__bf16)v.z; o[7] = (__bf16)v.w;
    reinterpret_cast<v8bf*>(dst)[i] = o;
}

// ---------------------------------------------------------------------------
// Fast GEMM (N % 128 == 0):  C[M,N] = act( A[M,K] * W[N,K]^T + bias )
// bf16 operands, f32 accumulate; software-pipelined K loop (fragments for
// k+32 load while k's WMMAs issue). block = 8 waves as 2x4; wave tile 32x32;
// block tile 64x128. Output f32 or bf16 (obf flag).
// ---------------------------------------------------------------------------
__global__ __launch_bounds__(256)
void gemm_wmma_fast(const __bf16* __restrict__ A, const __bf16* __restrict__ W,
                    const float* __restrict__ bias, void* __restrict__ Cout,
                    int N, int K, int relu, int obf) {
    const int lane = threadIdx.x & 31;
    const int wave = threadIdx.x >> 5;
    const int wy = wave >> 2;       // 0..1
    const int wx = wave & 3;        // 0..3
    const int row0 = blockIdx.x * 64  + wy * 32;
    const int col0 = blockIdx.y * 128 + wx * 32;

    v8f acc00 = {0.f,0.f,0.f,0.f,0.f,0.f,0.f,0.f};
    v8f acc01 = acc00, acc10 = acc00, acc11 = acc00;

    v16bf a0 = frag_a_bf16(A, K, row0,      0, lane);
    v16bf a1 = frag_a_bf16(A, K, row0 + 16, 0, lane);
    v16bf b0 = frag_b_bf16(W, K, col0,      0, lane);
    v16bf b1 = frag_b_bf16(W, K, col0 + 16, 0, lane);

    int k = 0;
    for (; k + 32 < K; k += 32) {
        v16bf na0 = frag_a_bf16(A, K, row0,      k + 32, lane);
        v16bf na1 = frag_a_bf16(A, K, row0 + 16, k + 32, lane);
        v16bf nb0 = frag_b_bf16(W, K, col0,      k + 32, lane);
        v16bf nb1 = frag_b_bf16(W, K, col0 + 16, k + 32, lane);
        acc00 = wmma_bf16(a0, b0, acc00);
        acc01 = wmma_bf16(a0, b1, acc01);
        acc10 = wmma_bf16(a1, b0, acc10);
        acc11 = wmma_bf16(a1, b1, acc11);
        a0 = na0; a1 = na1; b0 = nb0; b1 = nb1;
    }
    acc00 = wmma_bf16(a0, b0, acc00);
    acc01 = wmma_bf16(a0, b1, acc01);
    acc10 = wmma_bf16(a1, b0, acc10);
    acc11 = wmma_bf16(a1, b1, acc11);

    const int mb  = (lane >> 4) << 3;
    const int nc0 = col0 + (lane & 15);
    const float bv0 = bias ? bias[nc0]      : 0.0f;
    const float bv1 = bias ? bias[nc0 + 16] : 0.0f;
    float*  Cf = (float*)Cout;
    __bf16* Cb = (__bf16*)Cout;
    #pragma unroll
    for (int r = 0; r < 8; ++r) {
        const size_t rowA = (size_t)(row0 + mb + r) * N;
        const size_t rowB = (size_t)(row0 + 16 + mb + r) * N;
        float v00 = acc00[r] + bv0, v01 = acc01[r] + bv1;
        float v10 = acc10[r] + bv0, v11 = acc11[r] + bv1;
        if (relu) {
            v00 = fmaxf(v00, 0.0f); v01 = fmaxf(v01, 0.0f);
            v10 = fmaxf(v10, 0.0f); v11 = fmaxf(v11, 0.0f);
        }
        if (obf) {
            Cb[rowA + nc0]      = (__bf16)v00; Cb[rowA + nc0 + 16] = (__bf16)v01;
            Cb[rowB + nc0]      = (__bf16)v10; Cb[rowB + nc0 + 16] = (__bf16)v11;
        } else {
            Cf[rowA + nc0]      = v00; Cf[rowA + nc0 + 16] = v01;
            Cf[rowB + nc0]      = v10; Cf[rowB + nc0 + 16] = v11;
        }
    }
}

// ---------------------------------------------------------------------------
// Guarded GEMM (ragged N, 6-class head). bf16 in, f32 out. Wave tile 16x32.
// ---------------------------------------------------------------------------
__global__ __launch_bounds__(256)
void gemm_wmma_guard(const __bf16* __restrict__ A, const __bf16* __restrict__ W,
                     const float* __restrict__ bias, float* __restrict__ C,
                     int N, int K, int relu) {
    const int lane = threadIdx.x & 31;
    const int wave = threadIdx.x >> 5;
    const int wy = wave & 3;        // 0..3
    const int wx = wave >> 2;       // 0..1
    const int row0 = blockIdx.x * 64 + wy * 16;
    const int col0 = blockIdx.y * 64 + wx * 32;

    v8f acc0 = {0.f,0.f,0.f,0.f,0.f,0.f,0.f,0.f};
    v8f acc1 = acc0;

    for (int k = 0; k < K; k += 32) {
        v16bf a  = frag_a_bf16(A, K, row0, k, lane);
        v16bf b0 = frag_b_bf16_guard(W, K, col0,      k, N, lane);
        v16bf b1 = frag_b_bf16_guard(W, K, col0 + 16, k, N, lane);
        acc0 = wmma_bf16(a, b0, acc0);
        acc1 = wmma_bf16(a, b1, acc1);
    }

    const int mb = (lane >> 4) << 3;
    #pragma unroll
    for (int t = 0; t < 2; ++t) {
        const v8f acc = t ? acc1 : acc0;
        const int n = col0 + t * 16 + (lane & 15);
        if (n >= N) continue;
        const float bv = bias ? bias[n] : 0.0f;
        #pragma unroll
        for (int r = 0; r < 8; ++r) {
            float v = acc[r] + bv;
            if (relu) v = fmaxf(v, 0.0f);
            C[(size_t)(row0 + mb + r) * N + n] = v;
        }
    }
}

// ---------------------------------------------------------------------------
// Attention: one wave per (b, head, 16-query tile). qkv is bf16.
//   S = (Q K^T) * scale  (WMMA, into f32 LDS)
//   softmax -> * pe -> renormalize (+1e-6)      [fp32]
//   O = A V (WMMA, V staged through bf16 LDS), * degree, bf16 store
// ---------------------------------------------------------------------------
#define SC_LD 513
#define VT_LD 80    // bf16 elements per row (64 data + pad); 160B, 16B aligned

__global__ __launch_bounds__(32)
void attn_kernel(const __bf16* __restrict__ qkv,      // (B*N, 3D) bf16
                 const float* __restrict__ pe,        // (B, N, N)
                 const unsigned char* __restrict__ masks, // (B, N) bool
                 const float* __restrict__ degree,    // (B, N)
                 __bf16* __restrict__ obuf) {         // (B*N, D) bf16
    const int lane = threadIdx.x;
    const int bh = blockIdx.y;
    const int b  = bh / NHEAD;
    const int hh = bh % NHEAD;
    const int row0 = blockIdx.x * 16;                 // query tile base

    __shared__ float  sc[16 * SC_LD];
    __shared__ __bf16 vt[32 * VT_LD];

    const __bf16* base = qkv + (size_t)b * NN * TD;
    const float scale = 0.125f;                       // 1/sqrt(64)

    // ---- Q fragments (K-dim = DH = 64 -> 2 steps of 32) ----
    v16bf qf0 = frag_a_bf16(base + hh * DH, TD, row0, 0,  lane);
    v16bf qf1 = frag_a_bf16(base + hh * DH, TD, row0, 32, lane);

    // ---- scores: 32 key tiles of 16 ----
    const int mb = (lane >> 4) << 3;
    for (int nt = 0; nt < 32; ++nt) {
        v8f s = {0.f,0.f,0.f,0.f,0.f,0.f,0.f,0.f};
        v16bf kf0 = frag_b_bf16(base + DMODEL + hh * DH, TD, nt * 16, 0,  lane);
        v16bf kf1 = frag_b_bf16(base + DMODEL + hh * DH, TD, nt * 16, 32, lane);
        s = wmma_bf16(qf0, kf0, s);
        s = wmma_bf16(qf1, kf1, s);
        const int n = nt * 16 + (lane & 15);
        const bool mk = masks[b * NN + n] != 0;
        #pragma unroll
        for (int r = 0; r < 8; ++r) {
            float v = mk ? -1.0e9f : s[r] * scale;
            sc[(mb + r) * SC_LD + n] = v;
        }
    }
    __builtin_amdgcn_s_barrier();

    // ---- softmax * pe, renormalized. lane pair (row, half) each scans 256 ----
    {
        const int row  = lane & 15;
        const int half = lane >> 4;
        float* r = sc + row * SC_LD + half * 256;
        float mx = -3.0e38f;
        for (int j = 0; j < 256; ++j) mx = fmaxf(mx, r[j]);
        mx = fmaxf(mx, __shfl_xor(mx, 16, 32));
        float esum = 0.0f;
        for (int j = 0; j < 256; ++j) { float e = __expf(r[j] - mx); r[j] = e; esum += e; }
        esum += __shfl_xor(esum, 16, 32);
        const float* perow = pe + ((size_t)b * NN + (row0 + row)) * NN + half * 256;
        float psum = 0.0f;
        for (int j = 0; j < 256; ++j) { float w = r[j] * perow[j]; r[j] = w; psum += w; }
        psum += __shfl_xor(psum, 16, 32);
        // final_j = e_j*pe_j / (sum(e*pe) + 1e-6 * sum(e))
        const float inv = 1.0f / (psum + 1.0e-6f * esum);
        for (int j = 0; j < 256; ++j) r[j] *= inv;
    }
    __builtin_amdgcn_s_barrier();

    // ---- O = A(16x512) * V(512x64) ----
    v8f oa0 = {0.f,0.f,0.f,0.f,0.f,0.f,0.f,0.f};
    v8f oa1 = oa0, oa2 = oa0, oa3 = oa0;

    for (int ks = 0; ks < 16; ++ks) {
        const int kk = ks * 32;
        // stage 32 V rows (DH=64 bf16) into LDS; lane copies one row (8x16B)
        {
            const v8bf* vr = reinterpret_cast<const v8bf*>(
                base + (size_t)(kk + lane) * TD + 2 * DMODEL + hh * DH);
            v8bf* dst = reinterpret_cast<v8bf*>(vt + lane * VT_LD);
            #pragma unroll
            for (int i = 0; i < 8; ++i) dst[i] = vr[i];
        }
        __builtin_amdgcn_s_barrier();

        // A fragment from LDS scores (f32 -> bf16)
        v16bf af;
        {
            const int m  = lane & 15;
            const int kb = (lane >> 4) << 4;
            const float* p = sc + m * SC_LD + kk + kb;
            #pragma unroll
            for (int i = 0; i < 8; ++i) {
                af[i]     = (__bf16)p[i];
                af[8 + i] = (__bf16)p[8 + i];
            }
        }
        #pragma unroll
        for (int dt = 0; dt < 4; ++dt) {
            v16bf bf;
            const int d  = dt * 16 + (lane & 15);
            const int rb = (lane >> 4) << 4;
            #pragma unroll
            for (int i = 0; i < 16; ++i) bf[i] = vt[(rb + i) * VT_LD + d];
            v8f* acc = dt == 0 ? &oa0 : dt == 1 ? &oa1 : dt == 2 ? &oa2 : &oa3;
            *acc = wmma_bf16(af, bf, *acc);
        }
        __builtin_amdgcn_s_barrier();
    }

    // ---- store O * degree (bf16) ----
    #pragma unroll
    for (int dt = 0; dt < 4; ++dt) {
        const v8f acc = dt == 0 ? oa0 : dt == 1 ? oa1 : dt == 2 ? oa2 : oa3;
        const int d = dt * 16 + (lane & 15);
        #pragma unroll
        for (int r = 0; r < 8; ++r) {
            const int qn = row0 + mb + r;
            const float deg = degree[b * NN + qn];
            obuf[((size_t)b * NN + qn) * DMODEL + hh * DH + d] = (__bf16)(acc[r] * deg);
        }
    }
}

// ---------------------------------------------------------------------------
// h = LayerNorm(h + resid); writes f32 h and bf16 mirror hb. Rows of 512.
// ---------------------------------------------------------------------------
__global__ __launch_bounds__(256)
void add_ln_kernel(float* __restrict__ h, const float* __restrict__ resid,
                   const float* __restrict__ gamma, const float* __restrict__ beta,
                   __bf16* __restrict__ hb) {
    const int row = blockIdx.x;
    const int t = threadIdx.x;
    float* hr = h + (size_t)row * DMODEL;
    const float* rr = resid + (size_t)row * DMODEL;

    float x0 = hr[t] + rr[t];
    float x1 = hr[t + 256] + rr[t + 256];
    float s = x0 + x1;
    float q = x0 * x0 + x1 * x1;
    #pragma unroll
    for (int off = 16; off > 0; off >>= 1) {
        s += __shfl_xor(s, off, 32);
        q += __shfl_xor(q, off, 32);
    }
    __shared__ float ss[8], sq[8];
    const int wv = t >> 5, ln = t & 31;
    if (ln == 0) { ss[wv] = s; sq[wv] = q; }
    __syncthreads();
    if (t == 0) {
        float S = 0.f, Q = 0.f;
        #pragma unroll
        for (int i = 0; i < 8; ++i) { S += ss[i]; Q += sq[i]; }
        ss[0] = S; sq[0] = Q;
    }
    __syncthreads();
    const float mean = ss[0] * (1.0f / DMODEL);
    const float var  = sq[0] * (1.0f / DMODEL) - mean * mean;
    const float rstd = rsqrtf(var + 1.0e-5f);
    const float y0 = (x0 - mean) * rstd * gamma[t]       + beta[t];
    const float y1 = (x1 - mean) * rstd * gamma[t + 256] + beta[t + 256];
    hr[t] = y0;  hr[t + 256] = y1;
    __bf16* hbr = hb + (size_t)row * DMODEL;
    hbr[t] = (__bf16)y0;  hbr[t + 256] = (__bf16)y1;
}

// ---------------------------------------------------------------------------
// Launch
// ---------------------------------------------------------------------------
static inline void cvt_async(const float* s, __bf16* d, size_t count, hipStream_t st) {
    int n8 = (int)(count / 8);
    cvt_bf16_kernel<<<dim3((n8 + 255) / 256), dim3(256), 0, st>>>(s, d, n8);
}

extern "C" void kernel_launch(void* const* d_in, const int* in_sizes, int n_in,
                              void* d_out, int out_size, void* d_ws, size_t ws_size,
                              hipStream_t stream) {
    const float* x       = (const float*)d_in[0];
    const unsigned char* masks = (const unsigned char*)d_in[4];
    const float* pe      = (const float*)d_in[5];
    const float* degree  = (const float*)d_in[6];
    const float* W_emb   = (const float*)d_in[7];
    const float* qkv_w   = (const float*)d_in[8];
    const float* qkv_b   = (const float*)d_in[9];
    const float* out_w   = (const float*)d_in[10];
    const float* out_b   = (const float*)d_in[11];
    const float* lin1_w  = (const float*)d_in[12];
    const float* lin1_b  = (const float*)d_in[13];
    const float* lin2_w  = (const float*)d_in[14];
    const float* lin2_b  = (const float*)d_in[15];
    const float* ln1_s   = (const float*)d_in[16];
    const float* ln1_b   = (const float*)d_in[17];
    const float* ln2_s   = (const float*)d_in[18];
    const float* ln2_b   = (const float*)d_in[19];
    const float* cls1_w  = (const float*)d_in[20];
    const float* cls1_b  = (const float*)d_in[21];
    const float* cls2_w  = (const float*)d_in[22];
    const float* cls2_b  = (const float*)d_in[23];
    float* out = (float*)d_out;

    // ---- workspace layout (bytes) ----
    char* ws = (char*)d_ws;
    float*  h     = (float*) (ws);                      // 32 MB f32
    float*  tmp   = (float*) (ws + 33554432ull);        // 32 MB f32
    __bf16* hb    = (__bf16*)(ws + 67108864ull);        // 16 MB bf16
    __bf16* shrd  = (__bf16*)(ws + 83886080ull);        // 64 MB: qkvb / ff1b / clsb
    __bf16* obufb = (__bf16*)(ws + 150994944ull);       // 16 MB bf16
    __bf16* wpool = (__bf16*)(ws + 167772160ull);       // bf16 weights (~25.8 MB)
    __bf16* qkvb  = shrd;                               // 16384*1536 bf16
    __bf16* ff1b  = shrd;                               // 16384*2048 bf16
    __bf16* clsb  = shrd;                               // 16384*512 bf16

    // bf16 weight pool offsets (elements)
    __bf16* W_embb  = wpool;                                     // 512*64
    __bf16* qkv_wb  = W_embb  + 512ull * 64;                     // 4*1536*512
    __bf16* out_wb  = qkv_wb  + 4ull * TD * DMODEL;              // 4*512*512
    __bf16* lin1_wb = out_wb  + 4ull * DMODEL * DMODEL;          // 4*2048*512
    __bf16* lin2_wb = lin1_wb + 4ull * DFF * DMODEL;             // 4*512*2048
    __bf16* cls1_wb = lin2_wb + 4ull * DMODEL * DFF;             // 512*512
    __bf16* cls2_wb = cls1_wb + 512ull * 512;                    // 6*512
    __bf16* xb      = cls2_wb + 8ull * 512;                      // 16384*64 (pad to 8 rows)

    // ---- one-shot f32 -> bf16 packing of weights and x ----
    cvt_async(W_emb,  W_embb,  512ull * 64,            stream);
    cvt_async(qkv_w,  qkv_wb,  4ull * TD * DMODEL,     stream);
    cvt_async(out_w,  out_wb,  4ull * DMODEL * DMODEL, stream);
    cvt_async(lin1_w, lin1_wb, 4ull * DFF * DMODEL,    stream);
    cvt_async(lin2_w, lin2_wb, 4ull * DMODEL * DFF,    stream);
    cvt_async(cls1_w, cls1_wb, 512ull * 512,           stream);
    cvt_async(cls2_w, cls2_wb, 6ull * 512,             stream);
    cvt_async(x,      xb,      (size_t)MROWS * INF_,   stream);

    const dim3 blk(256);

    // embedding: h = x @ W_emb^T (f32 out), then mirror to hb
    gemm_wmma_fast<<<dim3(MROWS / 64, DMODEL / 128), blk, 0, stream>>>(
        xb, W_embb, nullptr, h, DMODEL, INF_, 0, 0);
    cvt_async(h, hb, (size_t)MROWS * DMODEL, stream);

    for (int l = 0; l < NLAYER; ++l) {
        // qkv = h @ qkv_w^T + qkv_b   (bf16 out)
        gemm_wmma_fast<<<dim3(MROWS / 64, TD / 128), blk, 0, stream>>>(
            hb, qkv_wb + (size_t)l * TD * DMODEL, qkv_b + (size_t)l * TD,
            qkvb, TD, DMODEL, 0, 1);

        // attention (masked softmax * pe, renorm, *degree) -> bf16 obuf
        attn_kernel<<<dim3(NN / 16, BATCH * NHEAD), dim3(32), 0, stream>>>(
            qkvb, pe, masks, degree, obufb);

        // out projection (f32 out)
        gemm_wmma_fast<<<dim3(MROWS / 64, DMODEL / 128), blk, 0, stream>>>(
            obufb, out_wb + (size_t)l * DMODEL * DMODEL, out_b + (size_t)l * DMODEL,
            tmp, DMODEL, DMODEL, 0, 0);

        // h = LN(h + tmp) ; hb = bf16(h)
        add_ln_kernel<<<dim3(MROWS), blk, 0, stream>>>(
            h, tmp, ln1_s + (size_t)l * DMODEL, ln1_b + (size_t)l * DMODEL, hb);

        // ff1 = relu(h @ lin1_w^T + b)   (bf16 out)
        gemm_wmma_fast<<<dim3(MROWS / 64, DFF / 128), blk, 0, stream>>>(
            hb, lin1_wb + (size_t)l * DFF * DMODEL, lin1_b + (size_t)l * DFF,
            ff1b, DFF, DMODEL, 1, 1);

        // ff2 = ff1 @ lin2_w^T + b   (f32 out)
        gemm_wmma_fast<<<dim3(MROWS / 64, DMODEL / 128), blk, 0, stream>>>(
            ff1b, lin2_wb + (size_t)l * DMODEL * DFF, lin2_b + (size_t)l * DMODEL,
            tmp, DMODEL, DFF, 0, 0);

        // h = LN(h + tmp) ; hb = bf16(h)
        add_ln_kernel<<<dim3(MROWS), blk, 0, stream>>>(
            h, tmp, ln2_s + (size_t)l * DMODEL, ln2_b + (size_t)l * DMODEL, hb);
    }

    // classifier: relu(h @ cls1^T + b) (bf16) @ cls2^T + b -> out (f32)
    gemm_wmma_fast<<<dim3(MROWS / 64, DMODEL / 128), blk, 0, stream>>>(
        hb, cls1_wb, cls1_b, clsb, DMODEL, DMODEL, 1, 1);
    gemm_wmma_guard<<<dim3(MROWS / 64, 1), blk, 0, stream>>>(
        clsb, cls2_wb, cls2_b, out, NCLS, DMODEL, 0);
}
